// SparseEncoderIN_16758962389328
// MI455X (gfx1250) — matrix operand order repaired and checked by reading, not compiled, not run on hardware
//
#include <hip/hip_runtime.h>
#include <hip/hip_bf16.h>

// ---------- bf16 helpers (storage type: unsigned short) ----------
__device__ __forceinline__ unsigned short f2bf(float f) {
    unsigned int u = __float_as_uint(f);
    unsigned int r = u + 0x7fffu + ((u >> 16) & 1u);   // round-to-nearest-even
    return (unsigned short)(r >> 16);
}
__device__ __forceinline__ float bf2f(unsigned short h) {
    return __uint_as_float(((unsigned int)h) << 16);
}
__device__ __forceinline__ unsigned int pack_bf2(float a, float b) {
    return (unsigned int)f2bf(a) | ((unsigned int)f2bf(b) << 16);
}

// ---------- WMMA fragment types ----------
typedef __attribute__((ext_vector_type(16))) __bf16 v16bf;
typedef __attribute__((ext_vector_type(8)))  float  v8f;
typedef __attribute__((ext_vector_type(4)))  int    v4i;
union Frag { v16bf v; unsigned int u[8]; };

// ---------- CDNA5 async global->LDS copy (ASYNCcnt-tracked) ----------
#if __has_builtin(__builtin_amdgcn_global_load_async_to_lds_b128) && \
    __has_builtin(__builtin_amdgcn_s_wait_asynccnt)
#define USE_ASYNC 1
#define AS1 __attribute__((address_space(1)))
#define AS3 __attribute__((address_space(3)))
__device__ __forceinline__ void async_cp16(const unsigned short* g, unsigned short* l) {
    __builtin_amdgcn_global_load_async_to_lds_b128((AS1 v4i*)g, (AS3 v4i*)l, 0, 0);
}
__device__ __forceinline__ void async_wait() { __builtin_amdgcn_s_wait_asynccnt(0); }
#else
#define USE_ASYNC 0
__device__ __forceinline__ void async_cp16(const unsigned short* g, unsigned short* l) {
    *reinterpret_cast<uint4*>(l) = *reinterpret_cast<const uint4*>(g);
}
__device__ __forceinline__ void async_wait() {}
#endif

// =====================================================================
// Layer 1: direct conv 3->64, stride 1, pad 1, D=64, with input masking.
// x NCDHW f32, mask int32; outputs h1 [B,64^3,64] bf16 (channels-last)
// and m1 [B,64^3] uint8 (dilated active mask).
// =====================================================================
__global__ __launch_bounds__(256)
void conv1_direct(const float* __restrict__ x,      // [4,3,262144]
                  const int*   __restrict__ mask,   // [4,262144]
                  const float* __restrict__ w1,     // [64,3,27]
                  const float* __restrict__ b1,     // [64]
                  unsigned short* __restrict__ h1,  // [4*262144,64]
                  unsigned char*  __restrict__ m1)  // [4*262144]
{
    __shared__ float lw[27 * 3 * 64];                // [t][ci][co]
    const int tid = threadIdx.x;
    for (int i = tid; i < 27 * 3 * 64; i += 256) {
        int co = i & 63; int rem = i >> 6; int ci = rem % 3; int t = rem / 3;
        lw[i] = w1[(co * 3 + ci) * 27 + t];
    }
    __syncthreads();

    const int v  = blockIdx.x * 256 + tid;           // 0 .. 4*262144-1
    const int xx = v & 63, yy = (v >> 6) & 63, zz = (v >> 12) & 63, b = v >> 18;

    float acc[64];
#pragma unroll
    for (int i = 0; i < 64; ++i) acc[i] = 0.f;
    bool act = false;
    const long vb = (long)b * 262144;

    for (int t = 0; t < 27; ++t) {
        const int tx = t % 3, ty = (t / 3) % 3, tz = t / 9;
        const int iz = zz - 1 + tz, iy = yy - 1 + ty, ix = xx - 1 + tx;
        if ((unsigned)iz < 64u && (unsigned)iy < 64u && (unsigned)ix < 64u) {
            const long vin = (long)((iz * 64 + iy) * 64 + ix);
            if (mask[vb + vin] < 1) {                // ACTIVE_THRESH = 1
                act = true;
#pragma unroll
                for (int ci = 0; ci < 3; ++ci) {
                    const float f = x[((long)(b * 3 + ci)) * 262144 + vin];
                    const float4* wv = reinterpret_cast<const float4*>(&lw[(t * 3 + ci) * 64]);
#pragma unroll
                    for (int q = 0; q < 16; ++q) {
                        float4 w4 = wv[q];
                        acc[4 * q + 0] += w4.x * f;
                        acc[4 * q + 1] += w4.y * f;
                        acc[4 * q + 2] += w4.z * f;
                        acc[4 * q + 3] += w4.w * f;
                    }
                }
            }
        }
    }

    m1[v] = act ? 1 : 0;
    unsigned int* hp = reinterpret_cast<unsigned int*>(h1) + (long)v * 32;
#pragma unroll
    for (int co = 0; co < 64; co += 2) {
        float v0 = act ? fmaxf(acc[co]     + b1[co],     0.f) : 0.f;
        float v1 = act ? fmaxf(acc[co + 1] + b1[co + 1], 0.f) : 0.f;
        hp[co >> 1] = pack_bf2(v0, v1);
    }
}

// =====================================================================
// Mask dilation: m_out[v] = OR over 3^3 stride-2 footprint of m_in.
// =====================================================================
template <int LOG2D>
__global__ __launch_bounds__(256)
void dilate_s2(const unsigned char* __restrict__ mi,
               unsigned char* __restrict__ mo)
{
    constexpr int D   = 1 << LOG2D;
    constexpr int Din = 2 * D;
    const int v  = blockIdx.x * 256 + threadIdx.x;
    const int xx = v & (D - 1);
    const int yy = (v >> LOG2D) & (D - 1);
    const int zz = (v >> (2 * LOG2D)) & (D - 1);
    const int b  = v >> (3 * LOG2D);
    if (b >= 4) return;
    bool act = false;
    for (int tz = 0; tz < 3; ++tz)
        for (int ty = 0; ty < 3; ++ty)
            for (int tx = 0; tx < 3; ++tx) {
                int iz = 2 * zz - 1 + tz, iy = 2 * yy - 1 + ty, ix = 2 * xx - 1 + tx;
                if ((unsigned)iz < (unsigned)Din && (unsigned)iy < (unsigned)Din &&
                    (unsigned)ix < (unsigned)Din)
                    act |= (mi[((long)(b * Din + iz) * Din + iy) * Din + ix] != 0);
            }
    mo[v] = act ? 1 : 0;
}

// =====================================================================
// Weight convert: f32 [Cout,Cin,27] -> bf16 [27,Cout,Cin]
// =====================================================================
__global__ __launch_bounds__(256)
void wconv(const float* __restrict__ src, unsigned short* __restrict__ dst,
           int Cout, int Cin, int total)
{
    int e = blockIdx.x * 256 + threadIdx.x;
    if (e >= total) return;
    int t = e % 27; int r = e / 27; int ci = r % Cin; int co = r / Cin;
    dst[((long)t * Cout + co) * Cin + ci] = f2bf(src[e]);
}

// =====================================================================
// Implicit-GEMM sparse conv block, stride 2, pad 1, k=3, bf16 WMMA.
//   hin  [B,Din^3,CIN] bf16 channels-last (zeros at inactive voxels)
//   wt   [27,COUT,CIN] bf16
//   mout [B,Dout^3] uint8 (already-dilated output mask)
//   hout [B,Dout^3,COUT] bf16
// Block = 256 threads = 8 waves. Block tile: VOX voxels x 128 couts;
// wave w owns cout tile blockIdx.y*128 + w*16 and VOX/16 accumulators,
// reusing one A fragment across all B tiles. Activations double-buffered
// in LDS, next tap staged with GLOBAL_LOAD_ASYNC_TO_LDS_B128 (ASYNCcnt)
// while the current tap's WMMAs run.
// =====================================================================
template <int CIN, int COUT, int LOG2D, int VOX>
__global__ __launch_bounds__(256)
void conv_wmma_s2(const unsigned short* __restrict__ hin,
                  const unsigned short* __restrict__ wt,
                  const float* __restrict__ bias,
                  const unsigned char* __restrict__ mout,
                  unsigned short* __restrict__ hout)
{
    constexpr int D    = 1 << LOG2D;       // output spatial dim
    constexpr int Din  = 2 * D;
    constexpr int NT   = VOX / 16;         // B tiles per wave
    constexpr int CPR  = CIN / 8;          // 16B chunks per voxel row
    constexpr int LCPR = __builtin_ctz(CPR);

    __shared__ __align__(16) unsigned short sBuf[2][VOX * CIN];

    const int tid  = threadIdx.x;
    const int lane = tid & 31;
    const int wave = tid >> 5;
    const int tile = blockIdx.x;                      // VOX-voxel tile id
    const int coutbase = blockIdx.y * 128 + wave * 16;

    const int hv   = lane >> 4;     // lane half (K-split per ISA layouts)
    const int nrow = lane & 15;     // A: M row; B/D: N column

    // ---- stage one tap's activation tile (VOX rows x CIN bf16) ----
    auto stage = [&](int buf, int t) {
        const int tx = t % 3, ty = (t / 3) % 3, tz = t / 9;
        for (int c = tid; c < VOX * CPR; c += 256) {
            const int n = c >> LCPR, q = c & (CPR - 1);
            const int vv = tile * VOX + n;
            const int xx = vv & (D - 1);
            const int yy = (vv >> LOG2D) & (D - 1);
            const int zz = (vv >> (2 * LOG2D)) & (D - 1);
            const int bb = vv >> (3 * LOG2D);
            const int iz = 2 * zz - 1 + tz;
            const int iy = 2 * yy - 1 + ty;
            const int ix = 2 * xx - 1 + tx;
            unsigned short* dst = &sBuf[buf][n * CIN + q * 8];
            if ((unsigned)iz < (unsigned)Din && (unsigned)iy < (unsigned)Din &&
                (unsigned)ix < (unsigned)Din) {
                const long vin = (long)((bb * Din + iz) * Din + iy) * Din + ix;
                async_cp16(hin + vin * CIN + q * 8, dst);   // ASYNCcnt path
            } else {
                *reinterpret_cast<uint4*>(dst) = make_uint4(0u, 0u, 0u, 0u);
            }
        }
    };

    v8f acc[NT];
#pragma unroll
    for (int nt = 0; nt < NT; ++nt) acc[nt] = v8f{};

    stage(0, 0);
    async_wait();
    __syncthreads();

    for (int t = 0; t < 27; ++t) {
        const int cur = t & 1;
        if (t < 26) stage(cur ^ 1, t + 1);            // overlap with WMMA below

        const unsigned short* wrow = wt + ((long)t * COUT + coutbase + nrow) * CIN;
        if (t < 26)                                   // warm next tap's weights
            __builtin_prefetch(wt + ((long)(t + 1) * COUT + coutbase + nrow) * CIN, 0, 3);

#pragma unroll
        for (int kc = 0; kc < CIN / 32; ++kc) {
            Frag a;
            // A 16x32 bf16: half 0 -> K 0..7 / 16..23, half 1 -> K 8..15 / 24..31
            *reinterpret_cast<uint4*>(&a.u[0]) =
                *reinterpret_cast<const uint4*>(wrow + kc * 32 + hv * 8);
            *reinterpret_cast<uint4*>(&a.u[4]) =
                *reinterpret_cast<const uint4*>(wrow + kc * 32 + 16 + hv * 8);
#pragma unroll
            for (int nt = 0; nt < NT; ++nt) {
                // B 32x16 bf16: half 0 -> K 0..7 / 16..23, half 1 -> K 8..15 / 24..31; N = lane%16
                const unsigned short* brow =
                    &sBuf[cur][(nt * 16 + nrow) * CIN + kc * 32 + hv * 8];
                Frag b;
                *reinterpret_cast<uint4*>(&b.u[0]) = *reinterpret_cast<const uint4*>(brow);
                *reinterpret_cast<uint4*>(&b.u[4]) = *reinterpret_cast<const uint4*>(brow + 16);
                acc[nt] = __builtin_amdgcn_wmma_f32_16x16x32_bf16(
                    false, a.v, false, b.v, (short)0, acc[nt], false, false);
            }
        }
        async_wait();
        __syncthreads();
    }

    // ---- epilogue: bias + ReLU where mask active, exact 0 elsewhere ----
    float bb8[8];
#pragma unroll
    for (int r = 0; r < 8; ++r) bb8[r] = bias[coutbase + hv * 8 + r];

#pragma unroll
    for (int nt = 0; nt < NT; ++nt) {
        const int vglob = tile * VOX + nt * 16 + nrow;     // this lane's voxel (N)
        const bool act = (mout[vglob] != 0);
        unsigned int packed[4];
#pragma unroll
        for (int r = 0; r < 8; r += 2) {
            float v0 = 0.f, v1 = 0.f;
            if (act) {
                v0 = fmaxf(acc[nt][r]     + bb8[r],     0.f);
                v1 = fmaxf(acc[nt][r + 1] + bb8[r + 1], 0.f);
            }
            packed[r >> 1] = pack_bf2(v0, v1);
        }
        const long obase = (long)vglob * COUT + coutbase + hv * 8;   // 16B aligned
        *reinterpret_cast<uint4*>(hout + obase) =
            make_uint4(packed[0], packed[1], packed[2], packed[3]);
    }
}

// =====================================================================
// Global max over 8^3 spatial + leaky_relu(0.2) -> f32 [4,512]
// =====================================================================
__global__ __launch_bounds__(256)
void maxpool_lrelu(const unsigned short* __restrict__ h4, float* __restrict__ out)
{
    int i = blockIdx.x * 256 + threadIdx.x;
    if (i >= 4 * 512) return;
    const int co = i % 512, b = i / 512;
    float m = -3.4e38f;
    for (int s = 0; s < 512; ++s)
        m = fmaxf(m, bf2f(h4[((long)(b * 512 + s)) * 512 + co]));
    out[i] = (m > 0.f) ? m : 0.2f * m;
}

// =====================================================================
extern "C" void kernel_launch(void* const* d_in, const int* in_sizes, int n_in,
                              void* d_out, int out_size, void* d_ws, size_t ws_size,
                              hipStream_t stream)
{
    const float* x    = (const float*)d_in[0];
    const int*   mask = (const int*)d_in[1];
    const float* w1   = (const float*)d_in[2];
    const float* b1   = (const float*)d_in[3];
    const float* w2   = (const float*)d_in[4];
    const float* b2   = (const float*)d_in[5];
    const float* w3   = (const float*)d_in[6];
    const float* b3   = (const float*)d_in[7];
    const float* w4   = (const float*)d_in[8];
    const float* b4   = (const float*)d_in[9];
    float* out = (float*)d_out;

    char* ws = (char*)d_ws;
    size_t off = 0;
    auto alloc = [&](size_t bytes) -> char* {
        char* p = ws + off;
        off = (off + bytes + 255) & ~(size_t)255;
        return p;
    };
    // bf16 activations (channels-last), uint8 masks, permuted bf16 weights
    unsigned short* h1  = (unsigned short*)alloc((size_t)4 * 262144 * 64 * 2);  // 134 MB
    unsigned short* h2  = (unsigned short*)alloc((size_t)4 * 32768 * 128 * 2);  //  33 MB
    unsigned short* h3  = (unsigned short*)alloc((size_t)4 * 4096 * 256 * 2);   // 8.4 MB
    unsigned short* h4  = (unsigned short*)alloc((size_t)4 * 512 * 512 * 2);    // 2.1 MB
    unsigned char*  m1  = (unsigned char*)alloc((size_t)4 * 262144);
    unsigned char*  m2  = (unsigned char*)alloc((size_t)4 * 32768);
    unsigned char*  m3  = (unsigned char*)alloc((size_t)4 * 4096);
    unsigned char*  m4  = (unsigned char*)alloc((size_t)4 * 512);
    unsigned short* wb2 = (unsigned short*)alloc((size_t)27 * 128 * 64 * 2);
    unsigned short* wb3 = (unsigned short*)alloc((size_t)27 * 256 * 128 * 2);
    unsigned short* wb4 = (unsigned short*)alloc((size_t)27 * 512 * 256 * 2);

    // weight permute/convert
    {
        int t2 = 128 * 64 * 27, t3 = 256 * 128 * 27, t4 = 512 * 256 * 27;
        wconv<<<(t2 + 255) / 256, 256, 0, stream>>>(w2, wb2, 128, 64, t2);
        wconv<<<(t3 + 255) / 256, 256, 0, stream>>>(w3, wb3, 256, 128, t3);
        wconv<<<(t4 + 255) / 256, 256, 0, stream>>>(w4, wb4, 512, 256, t4);
    }

    // layer 1 (direct) + mask m1
    conv1_direct<<<(4 * 262144) / 256, 256, 0, stream>>>(x, mask, w1, b1, h1, m1);

    // layer 2: 64 -> 128, 64^3 -> 32^3
    dilate_s2<5><<<(4 * 32768) / 256, 256, 0, stream>>>(m1, m2);
    conv_wmma_s2<64, 128, 5, 64><<<dim3((4 * 32768) / 64, 1), 256, 0, stream>>>(h1, wb2, b2, m2, h2);

    // layer 3: 128 -> 256, 32^3 -> 16^3
    dilate_s2<4><<<(4 * 4096) / 256, 256, 0, stream>>>(m2, m3);
    conv_wmma_s2<128, 256, 4, 64><<<dim3((4 * 4096) / 64, 2), 256, 0, stream>>>(h2, wb3, b3, m3, h3);

    // layer 4: 256 -> 512, 16^3 -> 8^3
    dilate_s2<3><<<(4 * 512) / 256, 256, 0, stream>>>(m3, m4);
    conv_wmma_s2<256, 512, 3, 32><<<dim3((4 * 512) / 32, 4), 256, 0, stream>>>(h3, wb4, b4, m4, h4);

    // global max-pool + leaky relu -> f32 [4,512]
    maxpool_lrelu<<<(2048 + 255) / 256, 256, 0, stream>>>(h4, out);
}
